// TemporalPropagator_13357348290704
// MI455X (gfx1250) — matrix operand order repaired and checked by reading, not compiled, not run on hardware
//
#include <hip/hip_runtime.h>

#define DD 1024
#define TBTOT 4096
#define BATCH 16
#define SPATIAL (TBTOT / BATCH)   // 256

typedef __attribute__((ext_vector_type(16))) __bf16 v16bf;
typedef __attribute__((ext_vector_type(8)))  float  v8f;

union Frag16 { v16bf v; unsigned int d[8]; };

__device__ __forceinline__ unsigned short f2bf(float f) {
  unsigned int x = __float_as_uint(f);
  x += 0x7FFFu + ((x >> 16) & 1u);   // round-to-nearest-even
  return (unsigned short)(x >> 16);
}
__device__ __forceinline__ unsigned int pk(float lo, float hi) {
  return (unsigned int)f2bf(lo) | ((unsigned int)f2bf(hi) << 16);
}
__device__ __forceinline__ uint4 pk8(float4 a, float4 b) {
  uint4 w;
  w.x = pk(a.x, a.y); w.y = pk(a.z, a.w);
  w.z = pk(b.x, b.y); w.w = pk(b.z, b.w);
  return w;
}

// ---------------------------------------------------------------------------
// Complex GEMM via 3-real-GEMM Karatsuba, bf16 WMMA, f32 accumulate.
//   P1 = Are*Bre ; P2 = Aim*Bim ; P3 = (Are+Aim)*(Bre+Bim)
//   Cre = P1 - P2 ; Cim = P3 - P1 - P2
// A: MxK row-major. transB==0: B KxN row-major. transB==1: B[k][n]=Bsrc[n*K+k]
// (C = A @ Bsrc^T). bimSign=-1 -> conjugate B. M,N,K multiples of 64 here.
//
// LDS tiles are FRAGMENT-MAJOR: 8 fragments (4 m/n-tiles x 2 k-tiles); each
// fragment = 32 lanes x 16 bf16 contiguous per lane (32 B), so fragment loads
// are 2x ds_load_b128 and staging stores are ds_store_b128. Three planes per
// side (re, im, re+im) -> zero VALU in the WMMA inner loop.
// ---------------------------------------------------------------------------
#define BM 64
#define BN 64
#define BK 64

__global__ __launch_bounds__(128)
void cgemm_wmma(const float* __restrict__ Are, const float* __restrict__ Aim,
                const float* __restrict__ Bre, const float* __restrict__ Bim,
                float* __restrict__ Cre, float* __restrict__ Cim,
                int M, int N, int K, int transB, float bimSign)
{
  // 6 planes x 8 fragments x 512 dwords = 96 KB
  __shared__ unsigned int sAre[8 * 512], sAim[8 * 512], sAsm[8 * 512];
  __shared__ unsigned int sBre[8 * 512], sBim[8 * 512], sBsm[8 * 512];

  const int tid  = threadIdx.x;
  const int lane = tid & 31;
  const int wave = tid >> 5;     // 0..3
  const int wr   = wave >> 1;    // wave row (0..1) -> 32 rows
  const int wc   = wave & 1;     // wave col (0..1) -> 32 cols
  const int half = lane >> 4;
  const int r    = lane & 15;

  const int m0 = blockIdx.x * BM;
  const int n0 = blockIdx.y * BN;

  v8f accP1[2][2], accP2[2][2], accP3[2][2];
  for (int a = 0; a < 2; a++)
    for (int b = 0; b < 2; b++)
      for (int i = 0; i < 8; i++) {
        accP1[a][b][i] = 0.f; accP2[a][b][i] = 0.f; accP3[a][b][i] = 0.f;
      }

  for (int kb = 0; kb < K; kb += BK) {
    // ---- stage A (64 rows x 64 k): thread takes 8 consecutive k of a row.
    // A-frag map: lane = half*16+row, element e = ebase + t <-> k = kk + t.
    for (int c = tid; c < 512; c += 128) {
      int row = c >> 3;                 // 0..63
      int k8  = (c & 7) * 8;            // 0..56
      int mtg = row >> 4, rr = row & 15;
      int kt  = k8 >> 5;
      int kk  = k8 & 31;                // 0,8,16,24
      int hf  = (kk >> 4) & 1;
      int edw = (kk & 8) >> 1;          // dword offset in lane: 0 or 4
      size_t g = (size_t)(m0 + row) * K + kb + k8;
      float4 r0 = ((const float4*)(Are + g))[0];
      float4 r1 = ((const float4*)(Are + g))[1];
      float4 i0 = ((const float4*)(Aim + g))[0];
      float4 i1 = ((const float4*)(Aim + g))[1];
      float4 s0, s1;
      s0.x = r0.x + i0.x; s0.y = r0.y + i0.y; s0.z = r0.z + i0.z; s0.w = r0.w + i0.w;
      s1.x = r1.x + i1.x; s1.y = r1.y + i1.y; s1.z = r1.z + i1.z; s1.w = r1.w + i1.w;
      int off = ((mtg * 2 + kt) << 9) + ((hf * 16 + rr) << 3) + edw;
      *(uint4*)&sAre[off] = pk8(r0, r1);
      *(uint4*)&sAim[off] = pk8(i0, i1);
      *(uint4*)&sAsm[off] = pk8(s0, s1);
    }
    // ---- stage B (64 n x 64 k): B-frag map: lane = half*16+n, e = k&15 ----
    if (transB) {
      // Bsrc row-major NxK: 8 consecutive k of one n -> contiguous global
      for (int c = tid; c < 512; c += 128) {
        int nn = c >> 3;
        int k8 = (c & 7) * 8;
        int ntg = nn >> 4, rr = nn & 15;
        int kt = k8 >> 5;
        int kk = k8 & 31;
        int hf  = (kk >> 4) & 1;
        int edw = (kk & 8) >> 1;        // 0 or 4
        size_t g = (size_t)(n0 + nn) * K + kb + k8;
        float4 r0 = ((const float4*)(Bre + g))[0];
        float4 r1 = ((const float4*)(Bre + g))[1];
        float4 i0 = ((const float4*)(Bim + g))[0];
        float4 i1 = ((const float4*)(Bim + g))[1];
        i0.x *= bimSign; i0.y *= bimSign; i0.z *= bimSign; i0.w *= bimSign;
        i1.x *= bimSign; i1.y *= bimSign; i1.z *= bimSign; i1.w *= bimSign;
        float4 s0, s1;
        s0.x = r0.x + i0.x; s0.y = r0.y + i0.y; s0.z = r0.z + i0.z; s0.w = r0.w + i0.w;
        s1.x = r1.x + i1.x; s1.y = r1.y + i1.y; s1.z = r1.z + i1.z; s1.w = r1.w + i1.w;
        int off = ((ntg * 2 + kt) << 9) + ((hf * 16 + rr) << 3) + edw;
        *(uint4*)&sBre[off] = pk8(r0, r1);
        *(uint4*)&sBim[off] = pk8(i0, i1);
        *(uint4*)&sBsm[off] = pk8(s0, s1);
      }
    } else {
      // Bsrc row-major KxN: 8 consecutive k at fixed n; consecutive tids
      // vary n -> strided reads coalesce across the wave.
      for (int c = tid; c < 512; c += 128) {
        int nn = c & 63;
        int k8 = (c >> 6) * 8;
        int ntg = nn >> 4, rr = nn & 15;
        int kt = k8 >> 5;
        int kk = k8 & 31;
        int hf  = (kk >> 4) & 1;
        int edw = (kk & 8) >> 1;
        const float* gr = Bre + (size_t)(kb + k8) * N + n0 + nn;
        const float* gi = Bim + (size_t)(kb + k8) * N + n0 + nn;
        float4 r0, r1, i0, i1, s0, s1;
        r0.x = gr[0];          r0.y = gr[(size_t)1*N];
        r0.z = gr[(size_t)2*N]; r0.w = gr[(size_t)3*N];
        r1.x = gr[(size_t)4*N]; r1.y = gr[(size_t)5*N];
        r1.z = gr[(size_t)6*N]; r1.w = gr[(size_t)7*N];
        i0.x = bimSign * gi[0];          i0.y = bimSign * gi[(size_t)1*N];
        i0.z = bimSign * gi[(size_t)2*N]; i0.w = bimSign * gi[(size_t)3*N];
        i1.x = bimSign * gi[(size_t)4*N]; i1.y = bimSign * gi[(size_t)5*N];
        i1.z = bimSign * gi[(size_t)6*N]; i1.w = bimSign * gi[(size_t)7*N];
        s0.x = r0.x + i0.x; s0.y = r0.y + i0.y; s0.z = r0.z + i0.z; s0.w = r0.w + i0.w;
        s1.x = r1.x + i1.x; s1.y = r1.y + i1.y; s1.z = r1.z + i1.z; s1.w = r1.w + i1.w;
        int off = ((ntg * 2 + kt) << 9) + ((hf * 16 + rr) << 3) + edw;
        *(uint4*)&sBre[off] = pk8(r0, r1);
        *(uint4*)&sBim[off] = pk8(i0, i1);
        *(uint4*)&sBsm[off] = pk8(s0, s1);
      }
    }
    // prefetch next A stripe into L2 (global_prefetch_b8)
    if (kb + BK < K)
      __builtin_prefetch(&Are[(size_t)(m0 + (tid >> 1)) * K + kb + BK], 0, 0);
    __syncthreads();

    #pragma unroll
    for (int ks = 0; ks < BK; ks += 32) {
      const int kt = ks >> 5;
      Frag16 aRe[2], aIm[2], aSm[2];
      #pragma unroll
      for (int mt = 0; mt < 2; mt++) {
        int off = (((wr * 2 + mt) * 2 + kt) << 9) + (lane << 3);
        aRe[mt].v = *(const v16bf*)&sAre[off];   // 2x ds_load_b128 each
        aIm[mt].v = *(const v16bf*)&sAim[off];
        aSm[mt].v = *(const v16bf*)&sAsm[off];
      }
      #pragma unroll
      for (int nt = 0; nt < 2; nt++) {
        Frag16 bRe, bIm, bSm;
        int off = (((wc * 2 + nt) * 2 + kt) << 9) + (lane << 3);
        bRe.v = *(const v16bf*)&sBre[off];
        bIm.v = *(const v16bf*)&sBim[off];
        bSm.v = *(const v16bf*)&sBsm[off];
        #pragma unroll
        for (int mt = 0; mt < 2; mt++) {
          accP1[mt][nt] = __builtin_amdgcn_wmma_f32_16x16x32_bf16(
              false, aRe[mt].v, false, bRe.v, (short)0, accP1[mt][nt], false, false);
          accP2[mt][nt] = __builtin_amdgcn_wmma_f32_16x16x32_bf16(
              false, aIm[mt].v, false, bIm.v, (short)0, accP2[mt][nt], false, false);
          accP3[mt][nt] = __builtin_amdgcn_wmma_f32_16x16x32_bf16(
              false, aSm[mt].v, false, bSm.v, (short)0, accP3[mt][nt], false, false);
        }
      }
    }
    __syncthreads();
  }

  // Epilogue: Cre = P1 - P2 ; Cim = P3 - P1 - P2.
  // C 16x16 f32 per ISA layout: VGPR i -> row = i + 8*half, col = lane%16.
  for (int mt = 0; mt < 2; mt++)
    for (int nt = 0; nt < 2; nt++) {
      int crow0 = m0 + wr * 32 + mt * 16 + (half ? 8 : 0);
      int ccol  = n0 + wc * 32 + nt * 16 + r;
      #pragma unroll
      for (int i = 0; i < 8; i++) {
        float p1 = accP1[mt][nt][i], p2 = accP2[mt][nt][i], p3 = accP3[mt][nt][i];
        Cre[(size_t)(crow0 + i) * N + ccol] = p1 - p2;
        Cim[(size_t)(crow0 + i) * N + ccol] = p3 - p1 - p2;
      }
    }
}

// ---------------------------------------------------------------------------
// Elementwise helpers
// ---------------------------------------------------------------------------
__global__ void skew_prep(const float* __restrict__ ure, const float* __restrict__ uim,
                          const float* __restrict__ vre, const float* __restrict__ vim,
                          float* Aur, float* Aui, float* Avr, float* Avi)
{
  int idx = blockIdx.x * 256 + threadIdx.x;
  if (idx >= DD * DD) return;
  int rr = idx / DD, cc = idx % DD;
  int t = cc * DD + rr;
  Aur[idx] = 0.5f * (ure[idx] - ure[t]);
  Aui[idx] = 0.5f * (uim[idx] + uim[t]);
  Avr[idx] = 0.5f * (vre[idx] - vre[t]);
  Avi[idx] = 0.5f * (vim[idx] + vim[t]);
}

// S = I + 2A ; T = A   (Neumann seed for (I-A)^-1 (I+A) = I + 2*sum_k A^k)
__global__ void cayley_init(const float* Ar, const float* Ai,
                            float* Sr, float* Si, float* Tr, float* Ti)
{
  int idx = blockIdx.x * 256 + threadIdx.x;
  if (idx >= DD * DD) return;
  int rr = idx / DD, cc = idx % DD;
  float ar = Ar[idx], ai = Ai[idx];
  Sr[idx] = (rr == cc ? 1.f : 0.f) + 2.f * ar;
  Si[idx] = 2.f * ai;
  Tr[idx] = ar; Ti[idx] = ai;
}

// S += 2P ; T = P
__global__ void cayley_acc(const float* Pr, const float* Pi,
                           float* Sr, float* Si, float* Tr, float* Ti)
{
  int idx = blockIdx.x * 256 + threadIdx.x;
  if (idx >= DD * DD) return;
  Sr[idx] += 2.f * Pr[idx];
  Si[idx] += 2.f * Pi[idx];
  Tr[idx] = Pr[idx]; Ti[idx] = Pi[idx];
}

// Us = U * diag(exp(ls)) ; Vs = V * diag(exp(-ls))   (column scaling)
__global__ void svd_scale(const float* Ur, const float* Ui,
                          const float* Vr, const float* Vi,
                          const float* __restrict__ ls,
                          float* Usr, float* Usi, float* Vsr, float* Vsi)
{
  int idx = blockIdx.x * 256 + threadIdx.x;
  if (idx >= DD * DD) return;
  int cc = idx % DD;
  float s  = __expf(ls[cc]);
  float si = __expf(-ls[cc]);
  Usr[idx] = Ur[idx] * s;  Usi[idx] = Ui[idx] * s;
  Vsr[idx] = Vr[idx] * si; Vsi[idx] = Vi[idx] * si;
}

// out[b][o] = sum_k cat(a0,a1)[b][k] * W[o][k] + bias[o]   (B x 2D, K = 2D)
__global__ void small_mix(const float* __restrict__ a0, const float* __restrict__ a1,
                          const float* __restrict__ W, const float* __restrict__ bias,
                          float* __restrict__ out)
{
  int idx = blockIdx.x * 256 + threadIdx.x;
  if (idx >= BATCH * 2 * DD) return;
  int b = idx / (2 * DD), o = idx % (2 * DD);
  const float* w  = W + (size_t)o * 2 * DD;
  const float* x0 = a0 + (size_t)b * DD;
  const float* x1 = a1 + (size_t)b * DD;
  float acc = bias[o];
  for (int k = 0; k < DD; k++) acc += x0[k] * w[k];
  for (int k = 0; k < DD; k++) acc += x1[k] * w[DD + k];
  out[idx] = acc;
}

__global__ void flux_update(const float* __restrict__ xmix,
                            const float* __restrict__ fre, const float* __restrict__ fim,
                            const float* __restrict__ dre, const float* __restrict__ dim_,
                            float* fnr, float* fni, float* __restrict__ out_flux)
{
  int idx = blockIdx.x * 256 + threadIdx.x;
  if (idx >= BATCH * DD) return;
  int b = idx / DD, d = idx % DD;
  float xr = xmix[b * 2 * DD + d];
  float xi = xmix[b * 2 * DD + DD + d];
  float dr = 1.f / (1.f + __expf(-dre[d]));
  float di = dim_[d];
  float fr = fre[idx], fi = fim[idx];
  float nr = fr * dr - fi * di + xr;
  float ni = fr * di + fi * dr + xi;
  fnr[idx] = nr; fni[idx] = ni;
  out_flux[idx] = nr;                 // (2,B,D) real plane
  out_flux[BATCH * DD + idx] = ni;    // imag plane
}

__global__ void trans_ops(const float* __restrict__ ld, const float* __restrict__ lf,
                          const float* __restrict__ lawr, const float* __restrict__ lawi,
                          const float* __restrict__ dtp,
                          float* opdr, float* opdi, float* opfr, float* opfi)
{
  int d = blockIdx.x * 256 + threadIdx.x;
  if (d >= DD) return;
  float lr = -__expf(ld[d]) + lawr[d];
  float li = lf[d] + lawi[d];
  float sp = (-lr > 20.f) ? -lr : log1pf(__expf(-lr));   // softplus(-lr)
  float Lre = -sp, Lim = li;
  float dt_eff = dtp[0];               // DT_REF = 1
  float zr = Lre * dt_eff, zi = Lim * dt_eff;
  float ez = __expf(zr);
  float s = sinf(zi), c = cosf(zi);
  float er = ez * c, ei = ez * s;      // exp(Z)
  float mag = sqrtf(zr * zr + zi * zi);
  float p1r, p1i;
  if (mag < 1e-4f) {
    float z2r = zr * zr - zi * zi, z2i = 2.f * zr * zi;
    p1r = 1.f + 0.5f * zr + z2r * (1.f / 6.f);
    p1i = 0.5f * zi + z2i * (1.f / 6.f);
  } else {
    float nr = er - 1.f, ni = ei;
    float den = zr * zr + zi * zi;
    p1r = (nr * zr + ni * zi) / den;
    p1i = (ni * zr - nr * zi) / den;
  }
  opdr[d] = er; opdi[d] = ei;
  opfr[d] = p1r * dt_eff; opfi[d] = p1i * dt_eff;
}

// ht = ht*op_decay + (xt + src[b])*op_forcing  (complex, in place on ht)
__global__ void combine(float* __restrict__ htr, float* __restrict__ hti,
                        const float* __restrict__ xtr, const float* __restrict__ xti,
                        const float* __restrict__ outcat,
                        const float* __restrict__ opdr, const float* __restrict__ opdi,
                        const float* __restrict__ opfr, const float* __restrict__ opfi)
{
  int idx = blockIdx.x * 256 + threadIdx.x;
  if (idx >= TBTOT * DD) return;
  int row = idx / DD, d = idx % DD;
  int b = row / SPATIAL;
  float sr = outcat[b * 2 * DD + d];
  float si = outcat[b * 2 * DD + DD + d];
  float hr = htr[idx], hi = hti[idx];
  float xr = xtr[idx] + sr, xi = xti[idx] + si;
  float dr = opdr[d], di = opdi[d], fr = opfr[d], fi = opfi[d];
  float nr = hr * dr - hi * di + xr * fr - xi * fi;
  float ni = hr * di + hi * dr + xr * fi + xi * fr;
  htr[idx] = nr; hti[idx] = ni;
}

// ---------------------------------------------------------------------------
extern "C" void kernel_launch(void* const* d_in, const int* in_sizes, int n_in,
                              void* d_out, int out_size, void* d_ws, size_t ws_size,
                              hipStream_t stream)
{
  const float* h_re   = (const float*)d_in[0];
  const float* h_im   = (const float*)d_in[1];
  const float* x_re   = (const float*)d_in[2];
  const float* x_im   = (const float*)d_in[3];
  const float* xg_re  = (const float*)d_in[4];
  const float* xg_im  = (const float*)d_in[5];
  const float* fl_re  = (const float*)d_in[6];
  const float* fl_im  = (const float*)d_in[7];
  const float* dt     = (const float*)d_in[8];
  const float* uraw_r = (const float*)d_in[9];
  const float* uraw_i = (const float*)d_in[10];
  const float* vraw_r = (const float*)d_in[11];
  const float* vraw_i = (const float*)d_in[12];
  const float* lsig   = (const float*)d_in[13];
  const float* dec_r  = (const float*)d_in[14];
  const float* dec_i  = (const float*)d_in[15];
  const float* mix_w  = (const float*)d_in[16];
  const float* mix_b  = (const float*)d_in[17];
  const float* proj_w = (const float*)d_in[18];
  const float* proj_b = (const float*)d_in[19];
  const float* ld     = (const float*)d_in[20];
  const float* lf     = (const float*)d_in[21];
  const float* law_r  = (const float*)d_in[22];
  const float* law_i  = (const float*)d_in[23];

  float* ws = (float*)d_ws;
  const size_t SZ  = (size_t)DD * DD;       // 1M
  const size_t BIG = (size_t)TBTOT * DD;    // 4M

  float *Aur = ws + 0*SZ, *Aui = ws + 1*SZ, *Avr = ws + 2*SZ, *Avi = ws + 3*SZ;
  float *Ur  = ws + 4*SZ, *Ui  = ws + 5*SZ, *Vr  = ws + 6*SZ, *Vi  = ws + 7*SZ;
  float *Tr  = ws + 8*SZ, *Ti  = ws + 9*SZ, *Pr  = ws + 10*SZ, *Pi = ws + 11*SZ;
  // reuse after Cayley loops:
  float *Usr = Aur, *Usi = Aui, *Vsr = Avr, *Vsi = Avi;
  float *Mr  = Tr,  *Mi  = Ti,  *Mir = Pr,  *Mii = Pi;

  float *htr = ws + 12*SZ;
  float *hti = htr + BIG;
  float *xtr = hti + BIG;
  float *xti = xtr + BIG;
  float *xmix   = xti + BIG;                       // B*2D
  float *fnr    = xmix + (size_t)BATCH * 2 * DD;   // B*D
  float *fni    = fnr  + (size_t)BATCH * DD;
  float *outcat = fni  + (size_t)BATCH * DD;       // B*2D
  float *opdr   = outcat + (size_t)BATCH * 2 * DD;
  float *opdi   = opdr + DD;
  float *opfr   = opdi + DD;
  float *opfi   = opfr + DD;

  float* out = (float*)d_out;
  float* out_h    = out;               // (2,TB,1,D): re then im planes
  float* out_flux = out + 2 * BIG;     // (2,B,D)

  const int gEl = (DD * DD + 255) / 256;
  dim3 gDD(DD / BM, DD / BN);          // 16 x 16
  dim3 gTB(TBTOT / BM, DD / BN);       // 64 x 16

  // 1) skew-Hermitian parts
  skew_prep<<<gEl, 256, 0, stream>>>(uraw_r, uraw_i, vraw_r, vraw_i, Aur, Aui, Avr, Avi);

  // 2) Cayley via Neumann series: S = I + 2*sum_{k=1..6} A^k  (||A||~0.23)
  cayley_init<<<gEl, 256, 0, stream>>>(Aur, Aui, Ur, Ui, Tr, Ti);
  for (int k = 2; k <= 6; k++) {
    cgemm_wmma<<<gDD, 128, 0, stream>>>(Aur, Aui, Tr, Ti, Pr, Pi, DD, DD, DD, 0, 1.f);
    cayley_acc<<<gEl, 256, 0, stream>>>(Pr, Pi, Ur, Ui, Tr, Ti);
  }
  cayley_init<<<gEl, 256, 0, stream>>>(Avr, Avi, Vr, Vi, Tr, Ti);
  for (int k = 2; k <= 6; k++) {
    cgemm_wmma<<<gDD, 128, 0, stream>>>(Avr, Avi, Tr, Ti, Pr, Pi, DD, DD, DD, 0, 1.f);
    cayley_acc<<<gEl, 256, 0, stream>>>(Pr, Pi, Vr, Vi, Tr, Ti);
  }

  // 3) column scaling by exp(+-log_sigma)  (overwrites A buffers)
  svd_scale<<<gEl, 256, 0, stream>>>(Ur, Ui, Vr, Vi, lsig, Usr, Usi, Vsr, Vsi);

  // 4) M = Us @ V^H ; M_inv = Vs @ U^H   (transB=1, conj -> bimSign=-1)
  cgemm_wmma<<<gDD, 128, 0, stream>>>(Usr, Usi, Vr, Vi, Mr,  Mi,  DD, DD, DD, 1, -1.f);
  cgemm_wmma<<<gDD, 128, 0, stream>>>(Vsr, Vsi, Ur, Ui, Mir, Mii, DD, DD, DD, 1, -1.f);

  // 5) h_tilde = h @ M_inv^T ; x_tilde = x @ M_inv^T   (transB=1, no conj)
  cgemm_wmma<<<gTB, 128, 0, stream>>>(h_re, h_im, Mir, Mii, htr, hti, TBTOT, DD, DD, 1, 1.f);
  cgemm_wmma<<<gTB, 128, 0, stream>>>(x_re, x_im, Mir, Mii, xtr, xti, TBTOT, DD, DD, 1, 1.f);

  // 6) flux tracker
  int gMix = (BATCH * 2 * DD + 255) / 256;
  small_mix<<<gMix, 256, 0, stream>>>(xg_re, xg_im, mix_w, mix_b, xmix);
  flux_update<<<(BATCH * DD + 255) / 256, 256, 0, stream>>>(
      xmix, fl_re, fl_im, dec_r, dec_i, fnr, fni, out_flux);
  small_mix<<<gMix, 256, 0, stream>>>(fnr, fni, proj_w, proj_b, outcat);

  // 7) transition operators exp(Z), phi1(Z)
  trans_ops<<<(DD + 255) / 256, 256, 0, stream>>>(ld, lf, law_r, law_i, dt,
                                                  opdr, opdi, opfr, opfi);

  // 8) h_next_tilde = h_tilde*op_decay + (x_tilde + src)*op_forcing (in place)
  combine<<<(TBTOT * DD + 255) / 256, 256, 0, stream>>>(
      htr, hti, xtr, xti, outcat, opdr, opdi, opfr, opfi);

  // 9) h_out = h_next_tilde @ M^T  -> (re, im) planes of d_out
  cgemm_wmma<<<gTB, 128, 0, stream>>>(htr, hti, Mr, Mi, out_h, out_h + BIG,
                                      TBTOT, DD, DD, 1, 1.f);
}
